// SGAN_73469710565584
// MI455X (gfx1250) — matrix operand order repaired and checked by reading, not compiled, not run on hardware
//
#include <hip/hip_runtime.h>
#include <hip/hip_bf16.h>

typedef _Float16 half_t;
typedef __attribute__((ext_vector_type(16))) _Float16 v16h;
typedef __attribute__((ext_vector_type(8)))  _Float16 v8h;
typedef __attribute__((ext_vector_type(2)))  _Float16 v2h;
typedef __attribute__((ext_vector_type(8)))  float    v8f;

#define B_  16
#define C_  512
#define N_  1681      // 41*41
#define NP  1696      // N padded to multiple of 32 (106*16, 53*32)
#define CK  256
#define NT  106       // NP/16
#define EPSV 1e-5f

__device__ __forceinline__ v8f v8f_zero() {
    v8f z;
#pragma unroll
    for (int i = 0; i < 8; ++i) z[i] = 0.0f;
    return z;
}

// ---------------- kernel 0a: x (f32 [B][C][N]) -> Xh (f16 [B][C][NP]) and
// Xt (f16 [B][NP][C]), both zero-padded in n. 32x32 LDS tile transpose so both
// the x reads and the Xt writes are coalesced. Grid (53, 16, B), block 256.
__global__ void cvt_x_kernel(const float* __restrict__ x,
                             half_t* __restrict__ Xh, half_t* __restrict__ Xt) {
    __shared__ half_t tile[32 * 33];
    const int t = threadIdx.x;
    const int i = t >> 5, j = t & 31;          // i: 0..7, j: 0..31
    const int npb = blockIdx.x * 32;           // n-tile base (53 tiles -> NP)
    const int cb  = blockIdx.y * 32;           // c-tile base (16 tiles -> C)
    const int b   = blockIdx.z;

    // load + Xh write (coalesced over n)
#pragma unroll
    for (int r = 0; r < 4; ++r) {
        int cl = i + 8 * r;
        int np = npb + j;
        float v = (np < N_) ? x[((size_t)b * C_ + cb + cl) * N_ + np] : 0.0f;
        half_t h = (half_t)v;
        Xh[((size_t)b * C_ + cb + cl) * NP + np] = h;
        tile[cl * 33 + j] = h;
    }
    __syncthreads();
    // Xt write (coalesced over c)
#pragma unroll
    for (int r = 0; r < 4; ++r) {
        int nl = i + 8 * r;
        Xt[((size_t)b * NP + npb + nl) * C_ + cb + j] = tile[j * 33 + nl];
    }
}

// ---------------- kernel 0b: Wq/Wk f32 -> f16
__global__ void cvt_w_kernel(const float* __restrict__ Wq, const float* __restrict__ Wk,
                             half_t* __restrict__ Whq, half_t* __restrict__ Whk) {
    int idx = blockIdx.x * blockDim.x + threadIdx.x;
    const int per = CK * C_;
    if (idx < per)             Whq[idx] = (half_t)Wq[idx];
    else if (idx < 2 * per)    Whk[idx - per] = (half_t)Wk[idx - per];
}

// ---------------- kernel 1: projections, LDS-free.
// Block = 64 threads (2 waves). Wave0 -> Q, Wave1 -> K. Grid (NT, CK/16, B).
// D[16 ck][16 n] = W[16ck x 32c] * Xt^T[32c x 16n], K-loop over C=512.
// Both operands read contiguous f16 -> global_load_b128 clauses.
// Stores f16 as Qh/Khn [B][NP][CK] (row = n, contiguous in ck).
__global__ void proj_kernel(const half_t* __restrict__ Xt,
                            const half_t* __restrict__ Whq, const half_t* __restrict__ Whk,
                            const float* __restrict__ bq, const float* __restrict__ bk,
                            half_t* __restrict__ Qh, half_t* __restrict__ Khn) {
    const int t = threadIdx.x;
    const int lane = t & 31, wave = t >> 5;
    const int g = lane >> 4, lr = lane & 15;
    const int nb = blockIdx.x * 16, ckb = blockIdx.y * 16, b = blockIdx.z;

    const half_t* Wh   = wave ? Whk : Whq;
    const float*  bias = wave ? bk  : bq;
    half_t*       outp = wave ? Khn : Qh;

    const half_t* wrow = Wh + (size_t)(ckb + lr) * C_;          // A: row ck = lane%16
    const half_t* xrow = Xt + ((size_t)b * NP + nb + lr) * C_;  // B: col n  = lane%16

    v8f acc = v8f_zero();
#pragma unroll 4
    for (int kk = 0; kk < 16; ++kk) {
        const int cb = kk * 32;
        v16h a, bm;
#pragma unroll
        for (int u = 0; u < 8; ++u) {
            int k0 = (u < 4) ? (8 * g + 2 * u) : (16 + 8 * g + 2 * (u - 4));
            v2h pa = *(const v2h*)(wrow + cb + k0);
            a[2 * u] = pa[0]; a[2 * u + 1] = pa[1];
            int kb0 = (u < 4) ? (8 * g + 2 * u) : (16 + 8 * g + 2 * (u - 4));
            v2h pb = *(const v2h*)(xrow + cb + kb0);
            bm[2 * u] = pb[0]; bm[2 * u + 1] = pb[1];
        }
        acc = __builtin_amdgcn_wmma_f32_16x16x32_f16(false, a, false, bm,
                                                     (short)0, acc, false, false);
    }

    // D element: m(=ck_local) = 8g + r, n_local = lane%16. Add bias, store 8 f16 contiguous.
    const int n = nb + lr;
    half_t* dst = outp + ((size_t)b * NP + n) * CK + ckb + 8 * g;
    v8h ov;
#pragma unroll
    for (int r = 0; r < 8; ++r)
        ov[r] = (half_t)(acc[r] + bias[ckb + 8 * g + r]);
    *(v8h*)dst = ov;     // 16B-aligned
}

// ---------------- kernel 2: fused attention for one 16-row strip.
// Block = 256 threads (8 waves) -> 16 waves per WGP at 2 blocks/WGP (LDS 109KB).
// Grid (NT, B). LDS strip Sl[16][NP] f32 + reduction scratch (dynamic LDS).
__global__ void attn_kernel(const half_t* __restrict__ Qh, const half_t* __restrict__ Khn,
                            const half_t* __restrict__ Xh,
                            const float* __restrict__ fg, const float* __restrict__ bg,
                            const float* __restrict__ x, const float* __restrict__ gamma,
                            float* __restrict__ out) {
    extern __shared__ float smem[];
    float* Sl   = smem;                 // 16*NP
    float* red  = smem + 16 * NP;       // 256
    float* rmin = red + 256;            // 16
    float* rinv = rmin + 16;            // 16

    const int t = threadIdx.x;
    const int lane = t & 31, wave = t >> 5;          // 8 waves
    const int g = lane >> 4, lr = lane & 15;
    const int nb = blockIdx.x * 16, b = blockIdx.y;

    // Preload A (Q rows nb..nb+15, full Ck=256) once per wave: 8 k-slabs of 32
    v16h aq[8];
    {
        const half_t* qrow = Qh + ((size_t)b * NP + nb + lr) * CK;
#pragma unroll
        for (int kk = 0; kk < 8; ++kk) {
#pragma unroll
            for (int u = 0; u < 8; ++u) {
                int k0 = (u < 4) ? (8 * g + 2 * u) : (16 + 8 * g + 2 * (u - 4));
                v2h p = *(const v2h*)(qrow + kk * 32 + k0);
                aq[kk][2 * u] = p[0]; aq[kk][2 * u + 1] = p[1];
            }
        }
    }
    const float scale = 0.0625f;  // Ck^-0.5, Ck=256

    // ---- Phase A: S strip = (Q K^T) * scale * fg  (invalid cells -> +BIG)
    for (int j = wave; j < NT; j += 8) {
        const int mb = j * 16;
        const half_t* kbase = Khn + ((size_t)b * NP + mb + lr) * CK;
        // prefetch next tile's K row for this lane (512 B)
        if (j + 8 < NT) {
            const half_t* knext = kbase + (size_t)128 * CK;
            __builtin_prefetch(knext, 0, 1);
            __builtin_prefetch(knext + 128, 0, 1);
        }
        v8f acc = v8f_zero();
#pragma unroll
        for (int kk = 0; kk < 8; ++kk) {
            v16h bm;
#pragma unroll
            for (int u = 0; u < 8; ++u) {
                int k0 = (u < 4) ? (8 * g + 2 * u) : (16 + 8 * g + 2 * (u - 4));
                v2h p = *(const v2h*)(kbase + kk * 32 + k0);
                bm[2 * u] = p[0]; bm[2 * u + 1] = p[1];
            }
            acc = __builtin_amdgcn_wmma_f32_16x16x32_f16(false, aq[kk], false, bm,
                                                         (short)0, acc, false, false);
        }
        const int mg = mb + lr;
        const int mgc = (mg < N_) ? mg : (N_ - 1);
#pragma unroll
        for (int r = 0; r < 8; ++r) {
            int nl = 8 * g + r;
            int ng = nb + nl;
            int ngc = (ng < N_) ? ng : (N_ - 1);
            float f = fg[((size_t)b * N_ + ngc) * N_ + mgc];   // clamped, always in-bounds
            bool valid = (ng < N_) && (mg < N_);
            float s = valid ? (acc[r] * scale * f) : 1e30f;    // v_cndmask, no exec branch
            Sl[nl * NP + mg] = s;
        }
    }
    __syncthreads();

    // ---- Phase B: row-wise masked normalize.
    // 16 rows x 16 column-partitions: consecutive threads scan consecutive
    // columns of one row -> coalesced fg/bg reads (the dominant HBM stream).
    const int row = t >> 4, part = t & 15;
    const int ng = nb + row;
    const int ngc = (ng < N_) ? ng : (N_ - 1);
    const float* fgrow = fg + ((size_t)b * N_ + ngc) * N_;
    const float* bgrow = bg + ((size_t)b * N_ + ngc) * N_;

    float lmin = 1e30f;
    for (int m = part; m < N_; m += 16) lmin = fminf(lmin, Sl[row * NP + m]);
    red[t] = lmin;
    __syncthreads();
    if (t < 16) {
        float mn = red[t * 16];
#pragma unroll
        for (int p2 = 1; p2 < 16; ++p2) mn = fminf(mn, red[t * 16 + p2]);
        rmin[t] = mn;
    }
    __syncthreads();

    const float mnv = rmin[row];
    float lsum = 0.0f;
    for (int m = part; m < N_; m += 16) {
        float x1 = (Sl[row * NP + m] - mnv) * fgrow[m];
        Sl[row * NP + m] = x1;
        lsum += x1;
    }
    red[t] = lsum;
    __syncthreads();
    if (t < 16) {
        float sm = red[t * 16];
#pragma unroll
        for (int p2 = 1; p2 < 16; ++p2) sm += red[t * 16 + p2];
        rinv[t] = 1.0f / (sm + EPSV);
    }
    __syncthreads();

    const float iv = rinv[row];
    for (int m = part; m < N_; m += 16) Sl[row * NP + m] = Sl[row * NP + m] * iv + bgrow[m];
    for (int m = N_ + part; m < NP; m += 16) Sl[row * NP + m] = 0.0f;  // zero the pad cols
    __syncthreads();

    // ---- Phase C: context = P (16 x NP) * V (NP x C), fused gamma*ctx + x
    const float gam = gamma[0];
    for (int ct = wave; ct < 32; ct += 8) {
        const int cb = ct * 16;
        v8f acc = v8f_zero();
        const half_t* vrow = Xh + ((size_t)b * C_ + cb + lr) * NP;
        for (int mt = 0; mt < NP; mt += 32) {
            v16h am, bm;
#pragma unroll
            for (int u = 0; u < 8; ++u) {
                int k0 = (u < 4) ? (8 * g + 2 * u) : (16 + 8 * g + 2 * (u - 4));
                float2 pp = *(const float2*)(&Sl[lr * NP + mt + k0]);   // ds_load_b64
                am[2 * u] = (half_t)pp.x; am[2 * u + 1] = (half_t)pp.y;
                int kb0 = (u < 4) ? (8 * g + 2 * u) : (16 + 8 * g + 2 * (u - 4));
                v2h q = *(const v2h*)(vrow + mt + kb0);
                bm[2 * u] = q[0]; bm[2 * u + 1] = q[1];
            }
            acc = __builtin_amdgcn_wmma_f32_16x16x32_f16(false, am, false, bm,
                                                         (short)0, acc, false, false);
        }
        const int cg = cb + lr;
#pragma unroll
        for (int r = 0; r < 8; ++r) {
            int nl2 = 8 * g + r;
            int ng2 = nb + nl2;
            if (ng2 < N_) {
                size_t oi = ((size_t)b * C_ + cg) * N_ + ng2;
                out[oi] = gam * acc[r] + x[oi];
            }
        }
    }
}

extern "C" void kernel_launch(void* const* d_in, const int* in_sizes, int n_in,
                              void* d_out, int out_size, void* d_ws, size_t ws_size,
                              hipStream_t stream) {
    const float* x     = (const float*)d_in[0];
    const float* fg    = (const float*)d_in[1];
    const float* bg    = (const float*)d_in[2];
    const float* Wq    = (const float*)d_in[3];
    const float* bq    = (const float*)d_in[4];
    const float* Wk    = (const float*)d_in[5];
    const float* bk    = (const float*)d_in[6];
    const float* gamma = (const float*)d_in[7];
    float* out = (float*)d_out;

    // workspace carve-up (256B aligned)
    size_t off = 0;
    auto carve = [&](size_t bytes) -> size_t {
        size_t o = off;
        off = (off + bytes + 255) & ~(size_t)255;
        return o;
    };
    char* ws = (char*)d_ws;
    half_t* Xh  = (half_t*)(ws + carve((size_t)B_ * C_ * NP * sizeof(half_t)));
    half_t* Xt  = (half_t*)(ws + carve((size_t)B_ * NP * C_ * sizeof(half_t)));
    half_t* Whq = (half_t*)(ws + carve((size_t)CK * C_ * sizeof(half_t)));
    half_t* Whk = (half_t*)(ws + carve((size_t)CK * C_ * sizeof(half_t)));
    half_t* Qh  = (half_t*)(ws + carve((size_t)B_ * NP * CK * sizeof(half_t)));
    half_t* Khn = (half_t*)(ws + carve((size_t)B_ * NP * CK * sizeof(half_t)));

    // 0a: x -> f16 padded (both layouts), tiled transpose
    {
        dim3 grid(NP / 32, C_ / 32, B_);
        cvt_x_kernel<<<grid, 256, 0, stream>>>(x, Xh, Xt);
    }
    // 0b: weights -> f16
    {
        int total = 2 * CK * C_;
        cvt_w_kernel<<<(total + 255) / 256, 256, 0, stream>>>(Wq, Wk, Whq, Whk);
    }
    // 1: Q/K projections
    {
        dim3 grid(NT, CK / 16, B_);
        proj_kernel<<<grid, 64, 0, stream>>>(Xt, Whq, Whk, bq, bk, Qh, Khn);
    }
    // 2: fused attention
    {
        dim3 grid(NT, B_);
        size_t smem_bytes = (size_t)(16 * NP + 256 + 16 + 16) * sizeof(float); // ~109 KB
        attn_kernel<<<grid, 256, smem_bytes, stream>>>(Qh, Khn, Xh, fg, bg, x, gamma, out);
    }
}
